// SSN_16423954940397
// MI455X (gfx1250) — compile-verified, matrix operand order checked by reading, not run on hardware
//
#include <hip/hip_runtime.h>
#include <hip/hip_bf16.h>

typedef __attribute__((ext_vector_type(16))) _Float16 v16h;
typedef __attribute__((ext_vector_type(8)))  float    v8f;

#define BATCH 2
#define Hh 512
#define Ww 512
#define KH 32
#define KW 32
#define KK (KH*KW)
#define CELL 16
#define HW (Hh*Ww)
#define YX_SCALE 0.15625f      /* max(32/(0.4*512), 32/(0.4*512)) */
#define LAB_SCALE 0.26f
#define INVALID_DIST 1e10f

/* float offsets inside d_out (outputs concatenated in return order) */
#define OUT_PFEAT  0
#define OUT_SPFEAT (BATCH*5*HW)                  /* 2,621,440 */
#define OUT_ASSOC  (OUT_SPFEAT + BATCH*5*KK)     /* 2,631,680 */
#define OUT_FINAL  (OUT_ASSOC + BATCH*9*HW)      /* 7,350,272 */

/* -------------------------------------------------------------------- */
/* k0: compute pFeat, write it out, and compute initial per-cell means. */
__global__ void ssn_init(const float* __restrict__ img, float* out,
                         float* __restrict__ spf) {
    int cell = blockIdx.x;
    int b = cell / KK, k = cell % KK;
    int ki = k / KW, kj = k % KW;
    int t = threadIdx.x, py = t >> 4, px = t & 15;
    int y = ki*CELL + py, x = kj*CELL + px;

    float p[5];
    p[0] = YX_SCALE * (float)y;
    p[1] = YX_SCALE * (float)x;
#pragma unroll
    for (int c = 0; c < 3; ++c)
        p[2+c] = LAB_SCALE * img[((b*3 + c)*Hh + y)*Ww + x];
#pragma unroll
    for (int c = 0; c < 5; ++c)
        out[OUT_PFEAT + ((b*5 + c)*Hh + y)*Ww + x] = p[c];

    __shared__ float red[8][6];
    int lane = t & 31, wv = t >> 5;
#pragma unroll
    for (int c = 0; c < 5; ++c) {
        float v = p[c];
        for (int off = 16; off; off >>= 1) v += __shfl_down(v, off, 32);
        if (lane == 0) red[wv][c] = v;
    }
    __syncthreads();
    if (t < 5) {
        float s = 0.f;
#pragma unroll
        for (int w = 0; w < 8; ++w) s += red[w][t];
        spf[(b*KK + k)*8 + t] = s * (1.0f/256.0f);
    }
}

/* -------------------------------------------------------------------- */
/* kA: per-pixel distances to the 9 neighboring centroids via WMMA,     */
/* softmax, write assoc (and argmax -> final index; last call wins).    */
/* Fragments are full 32B LDS rows; lanes>=16 read a dedicated zero row */
/* (address select, not per-element data select).                       */
__global__ void ssn_assoc(float* out, const float* __restrict__ spf) {
    __shared__ _Float16 Ah[257][16];  /* rows 0..255 pixels, row 256 = 0 */
    __shared__ _Float16 Bh[17][16];   /* rows 0..8 centroids, 9..16 = 0  */
    __shared__ float    distL[256*16];
    __shared__ float    validL[16];
    __shared__ int      idxL[16];

    int cell = blockIdx.x;
    int b = cell / KK, k = cell % KK;
    int ki = k / KW, kj = k % KW;
    int t = threadIdx.x, py = t >> 4, px = t & 15;
    int y = ki*CELL + py, x = kj*CELL + px;

    /* tile-centering keeps f16 magnitudes small; distances unchanged */
    float cy = YX_SCALE * ((float)(ki*CELL) + 7.5f);
    float cx = YX_SCALE * ((float)(kj*CELL) + 7.5f);

    float p[5];
#pragma unroll
    for (int c = 0; c < 5; ++c)
        p[c] = out[OUT_PFEAT + ((b*5 + c)*Hh + y)*Ww + x];
    float pc0 = p[0] - cy, pc1 = p[1] - cx;
    float pn = pc0*pc0 + pc1*pc1 + p[2]*p[2] + p[3]*p[3] + p[4]*p[4];

    {   /* write full 32B row: aug features in K=0..5, zeros elsewhere */
        v16h arow = {};
        arow[0] = (_Float16)(-2.f*pc0);
        arow[1] = (_Float16)(-2.f*pc1);
        arow[2] = (_Float16)(-2.f*p[2]);
        arow[3] = (_Float16)(-2.f*p[3]);
        arow[4] = (_Float16)(-2.f*p[4]);
        arow[5] = (_Float16)1.0f;
        *(v16h*)&Ah[t][0] = arow;
    }
    if (t == 0) {                       /* zero row for lanes >= 16 */
        v16h z = {};
        *(v16h*)&Ah[256][0] = z;
    }
    if (t < 17) {
        v16h brow = {};
        if (t < 9) {
            int di = t/3 - 1, dj = t%3 - 1;
            int ni = ki + di, nj = kj + dj;
            bool ok = (ni >= 0) && (ni < KH) && (nj >= 0) && (nj < KW);
            int nic = min(max(ni, 0), KH-1), njc = min(max(nj, 0), KW-1);
            int nk = nic*KW + njc;
            const float* s = spf + (b*KK + nk)*8;
            float s0 = s[0]-cy, s1 = s[1]-cx, s2 = s[2], s3 = s[3], s4 = s[4];
            brow[0] = (_Float16)s0;
            brow[1] = (_Float16)s1;
            brow[2] = (_Float16)s2;
            brow[3] = (_Float16)s3;
            brow[4] = (_Float16)s4;
            brow[5] = (_Float16)(s0*s0 + s1*s1 + s2*s2 + s3*s3 + s4*s4);
            validL[t] = ok ? 1.f : 0.f;
            idxL[t]   = nk;
        } else if (t < 16) {
            validL[t] = 0.f;
            idxL[t]   = 0;
        }
        *(v16h*)&Bh[t][0] = brow;
    }
    __syncthreads();

    /* 8 waves x 2 M-tiles: D[m,n] = -2(p-c).(s-c) + |s-c|^2 */
    int lane = t & 31, wv = t >> 5;
    int l15 = lane & 15;
    bool lo = lane < 16;
    int arow_hi = 256, brow_hi = 16;    /* zero rows for the upper half */
    int brow = lo ? l15 : brow_hi;
#pragma unroll
    for (int tt = 0; tt < 2; ++tt) {
        int m = wv*2 + tt;
        int arow = lo ? (m*16 + l15) : arow_hi;
        v16h afrag = *(const v16h*)&Ah[arow][0];   /* 2x ds_load_b128 */
        v16h bfrag = *(const v16h*)&Bh[brow][0];
        v8f acc = {};
        acc = __builtin_amdgcn_wmma_f32_16x16x32_f16(
                  false, afrag, false, bfrag, (short)0, acc, false, false);
#pragma unroll
        for (int r = 0; r < 8; ++r) {
            int Mloc = lo ? r : (8 + r);  /* C/D: VGPR r <-> row r / r+8 */
            distL[(m*16 + Mloc)*16 + l15] = acc[r];
        }
    }
    __syncthreads();

    /* per-pixel softmax over the 9 neighbors */
    float d9[9];
    float dmin = 3.0e38f;
#pragma unroll
    for (int n = 0; n < 9; ++n) {
        float d = distL[t*16 + n] + pn;
        d = (validL[n] > 0.f) ? d : INVALID_DIST;
        d9[n] = d;
        dmin = fminf(dmin, d);
    }
    float e[9], sum = 0.f;
#pragma unroll
    for (int n = 0; n < 9; ++n) {
        e[n] = (validL[n] > 0.f) ? __expf(dmin - d9[n]) : 0.f;
        sum += e[n];
    }
    float inv = 1.f / sum;
    int best = 0; float bv = -1.f;
#pragma unroll
    for (int n = 0; n < 9; ++n) {
        float a = e[n] * inv;
        out[OUT_ASSOC + ((b*9 + n)*Hh + y)*Ww + x] = a;
        if (a > bv) { bv = a; best = n; }
    }
    out[OUT_FINAL + (b*Hh + y)*Ww + x] = (float)idxL[best];
}

/* -------------------------------------------------------------------- */
/* kB: gather-based centroid update (deterministic; no float atomics).  */
/* Prefetch all 9 association lines up front (global_prefetch_b8) to   */
/* hide L2 latency of the later cells behind the earlier cells' math.  */
__global__ void ssn_update(float* out, float* __restrict__ spf) {
    int cell = blockIdx.x;
    int b = cell / KK, k = cell % KK;
    int ki = k / KW, kj = k % KW;
    int t = threadIdx.x, py = t >> 4, px = t & 15;

#pragma unroll
    for (int di = -1; di <= 1; ++di) {
#pragma unroll
        for (int dj = -1; dj <= 1; ++dj) {
            int ni = ki + di, nj = kj + dj;
            if (ni < 0 || ni >= KH || nj < 0 || nj >= KW) continue;
            int slot = (1 - di)*3 + (1 - dj);
            int y = ni*CELL + py, x = nj*CELL + px;
            __builtin_prefetch(&out[OUT_ASSOC + ((b*9 + slot)*Hh + y)*Ww + x], 0, 0);
        }
    }

    float acc[6] = {0.f, 0.f, 0.f, 0.f, 0.f, 0.f};
    for (int di = -1; di <= 1; ++di) {
        for (int dj = -1; dj <= 1; ++dj) {
            int ni = ki + di, nj = kj + dj;
            if (ni < 0 || ni >= KH || nj < 0 || nj >= KW) continue;
            int slot = (1 - di)*3 + (1 - dj);  /* reciprocal neighbor slot */
            int y = ni*CELL + py, x = nj*CELL + px;
            float w = out[OUT_ASSOC + ((b*9 + slot)*Hh + y)*Ww + x];
#pragma unroll
            for (int c = 0; c < 5; ++c)
                acc[c] += w * out[OUT_PFEAT + ((b*5 + c)*Hh + y)*Ww + x];
            acc[5] += w;
        }
    }

    __shared__ float red[8][6];
    int lane = t & 31, wv = t >> 5;
#pragma unroll
    for (int c = 0; c < 6; ++c) {
        float v = acc[c];
        for (int off = 16; off; off >>= 1) v += __shfl_down(v, off, 32);
        if (lane == 0) red[wv][c] = v;
    }
    __syncthreads();
    if (t == 0) {
        float s[6];
#pragma unroll
        for (int c = 0; c < 6; ++c) {
            float v = 0.f;
#pragma unroll
            for (int w = 0; w < 8; ++w) v += red[w][c];
            s[c] = v;
        }
        float wsum = fmaxf(s[5], 1e-10f);
#pragma unroll
        for (int c = 0; c < 5; ++c) {
            float v = s[c] / wsum;
            spf[(b*KK + k)*8 + c] = v;                      /* for next iter */
            out[OUT_SPFEAT + (b*5 + c)*KK + k] = v;         /* last write = output */
        }
    }
}

/* -------------------------------------------------------------------- */
extern "C" void kernel_launch(void* const* d_in, const int* in_sizes, int n_in,
                              void* d_out, int out_size, void* d_ws, size_t ws_size,
                              hipStream_t stream) {
    (void)in_sizes; (void)n_in; (void)out_size; (void)ws_size;
    const float* img = (const float*)d_in[0];   /* img_lab (B,3,H,W) f32 */
    /* d_in[1] (init_spIndx) is the regular grid; recomputed analytically */
    float* out = (float*)d_out;
    float* spf = (float*)d_ws;                  /* B*K*8 floats = 64 KB  */

    dim3 grid(BATCH*KK), block(256);
    ssn_init<<<grid, block, 0, stream>>>(img, out, spf);
    for (int it = 1; it < 5; ++it) {
        ssn_assoc<<<grid, block, 0, stream>>>(out, spf);
        ssn_update<<<grid, block, 0, stream>>>(out, spf);
    }
    ssn_assoc<<<grid, block, 0, stream>>>(out, spf);
}
